// FiveLayerSAGE_80238579024178
// MI455X (gfx1250) — matrix operand order repaired and checked by reading, not compiled, not run on hardware
//
#include <hip/hip_runtime.h>

#define N_NODES 50000
#define N_EDGES 800000
#define IN_CH   128
#define HID     256
#define OUT_CH  10
#define N_GRAPHS 128

typedef __attribute__((ext_vector_type(16))) __bf16 v16bf;
typedef __attribute__((ext_vector_type(8)))  __bf16 v8bf;
typedef __attribute__((ext_vector_type(4)))  __bf16 v4bf;
typedef __attribute__((ext_vector_type(8)))  float  v8f;

// float -> bf16 round-to-nearest-even (bit manipulation; no cvt dependency)
static __device__ __forceinline__ __bf16 f2bf(float f) {
    unsigned u = __builtin_bit_cast(unsigned, f);
    unsigned short h = (unsigned short)((u + 0x7FFFu + ((u >> 16) & 1u)) >> 16);
    return __builtin_bit_cast(__bf16, h);
}
static __device__ __forceinline__ float bf2f(__bf16 b) {
    unsigned u = ((unsigned)__builtin_bit_cast(unsigned short, b)) << 16;
    return __builtin_bit_cast(float, u);
}
// 16 contiguous bf16 (two b128 loads)
static __device__ __forceinline__ v16bf ld16bf(const __bf16* p) {
    v8bf a = *(const v8bf*)p;
    v8bf b = *(const v8bf*)(p + 8);
    return __builtin_shufflevector(a, b, 0, 1, 2, 3, 4, 5, 6, 7,
                                         8, 9, 10, 11, 12, 13, 14, 15);
}
// A-fragment: two runs of 8 bf16 at +0 and +16 (16-bit A 16x32 layout)
static __device__ __forceinline__ v16bf ldA(const __bf16* p) {
    v8bf a = *(const v8bf*)p;
    v8bf b = *(const v8bf*)(p + 16);
    return __builtin_shufflevector(a, b, 0, 1, 2, 3, 4, 5, 6, 7,
                                         8, 9, 10, 11, 12, 13, 14, 15);
}

__global__ void zero_f32(float* __restrict__ p, long n) {
    long i = (long)blockIdx.x * blockDim.x + threadIdx.x;
    if (i < n) p[i] = 0.0f;
}

// vectorized f32 -> bf16 (n4 = count/4)
__global__ void f32_to_bf16_k(const float* __restrict__ src, __bf16* __restrict__ dst, long n4) {
    long i = (long)blockIdx.x * blockDim.x + threadIdx.x;
    if (i >= n4) return;
    const float4 f = ((const float4*)src)[i];
    v4bf o;
    o[0] = f2bf(f.x); o[1] = f2bf(f.y); o[2] = f2bf(f.z); o[3] = f2bf(f.w);
    ((v4bf*)dst)[i] = o;
}

// Pack W[CIN,256] (row-major f32) into WMMA B-fragment order bf16:
// P[((s*256 + n)*2 + hi)*16 + e] = W[(s*32 + hi*16 + e)*256 + n]
__global__ void pack_w(const float* __restrict__ W, __bf16* __restrict__ P, int cin) {
    int tid = blockIdx.x * blockDim.x + threadIdx.x;
    if (tid >= cin * HID) return;
    int e  = tid & 15;
    int hi = (tid >> 4) & 1;
    int n  = (tid >> 5) & 255;
    int s  = tid >> 13;
    int K  = s * 32 + hi * 16 + e;
    P[tid] = f2bf(W[K * HID + n]);
}

__global__ void deg_count(const int* __restrict__ ei, float* __restrict__ deg) {
    long e = (long)blockIdx.x * blockDim.x + threadIdx.x;
    if (e < N_EDGES) atomicAdd(&deg[ei[N_EDGES + e]], 1.0f);
}

__global__ void deg_invert(float* __restrict__ deg) {
    int i = blockIdx.x * blockDim.x + threadIdx.x;
    if (i < N_NODES) deg[i] = 1.0f / fmaxf(deg[i], 1.0f);
}

// layer-1 scatter: agg[dst,c] += x[src,c] * rdeg[dst]  (f32 src, 128 ch)
__global__ void edge_scatter_f32(const int* __restrict__ ei, const float* __restrict__ x,
                                 float* __restrict__ agg, const float* __restrict__ rdeg) {
    long tid = (long)blockIdx.x * blockDim.x + threadIdx.x;
    long e = tid >> 5;
    int  cg = (int)(tid & 31);
    if (e >= N_EDGES) return;
    int src = ei[e];
    int dst = ei[N_EDGES + e];
    float rd = rdeg[dst];
    const float4 v = *(const float4*)(x + (long)src * IN_CH + cg * 4);
    float* a = agg + (long)dst * IN_CH + cg * 4;
    atomicAdd(a + 0, v.x * rd);
    atomicAdd(a + 1, v.y * rd);
    atomicAdd(a + 2, v.z * rd);
    atomicAdd(a + 3, v.w * rd);
}

// layers 2..5 scatter: agg[dst,c] += h[src,c] * rdeg[dst]  (bf16 src, 256 ch)
__global__ void edge_scatter_bf16(const int* __restrict__ ei, const __bf16* __restrict__ h,
                                  float* __restrict__ agg, const float* __restrict__ rdeg) {
    long tid = (long)blockIdx.x * blockDim.x + threadIdx.x;
    long e = tid >> 6;
    int  cg = (int)(tid & 63);
    if (e >= N_EDGES) return;
    int src = ei[e];
    int dst = ei[N_EDGES + e];
    float rd = rdeg[dst];
    const v4bf v = *(const v4bf*)(h + (long)src * HID + cg * 4);
    float* a = agg + (long)dst * HID + cg * 4;
    atomicAdd(a + 0, bf2f(v[0]) * rd);
    atomicAdd(a + 1, bf2f(v[1]) * rd);
    atomicAdd(a + 2, bf2f(v[2]) * rd);
    atomicAdd(a + 3, bf2f(v[3]) * rd);
}

// out = relu( mean @ Wl + hx @ Wr + bl ) -> bf16 row-major (may be in place over hx)
// Block = 16 nodes x 256 out-cols, 8 waves; wave w -> N-tiles {16w, 16w+128}.
// 4 independent WMMA accumulator chains per wave (avoids bf16 WMMA RAW stalls).
template<int CIN>
__global__ __launch_bounds__(256)
void sage_gemm(const __bf16* __restrict__ meanbf, // [N, CIN] bf16 (scaled by rdeg)
               const __bf16* __restrict__ hx,     // [N, CIN] bf16
               const __bf16* __restrict__ pWl,    // packed B fragments
               const __bf16* __restrict__ pWr,
               const float* __restrict__ bl,      // [HID]
               __bf16* __restrict__ out)          // [N, HID] bf16 (may alias hx)
{
    const int lane = threadIdx.x & 31;
    const int wave = threadIdx.x >> 5;          // 0..7
    const int hi   = lane >> 4;
    const int mrow = lane & 15;
    const int m0   = blockIdx.x * 16;
    const int node = m0 + mrow;
    const int n0   = wave * 16 + mrow;          // first output column
    const int n1   = n0 + 128;                  // second output column
    const __bf16* xrow = hx     + (long)node * CIN;
    const __bf16* arow = meanbf + (long)node * CIN;

    v8f cm0 = {}, cx0 = {}, cm1 = {}, cx1 = {};
    for (int s = 0; s < CIN / 32; ++s) {
        const int ka = s * 32 + hi * 16;
        const v16bf am = ld16bf(arow + ka);
        const v16bf ax = ld16bf(xrow + ka);
        const long b0 = ((long)(s * 256 + n0) * 2 + hi) * 16;
        const long b1 = ((long)(s * 256 + n1) * 2 + hi) * 16;
        const v16bf bL0 = ld16bf(pWl + b0);
        const v16bf bR0 = ld16bf(pWr + b0);
        const v16bf bL1 = ld16bf(pWl + b1);
        const v16bf bR1 = ld16bf(pWr + b1);
        cm0 = __builtin_amdgcn_wmma_f32_16x16x32_bf16(false, am, false, bL0,
                                                      (short)0, cm0, false, false);
        cx0 = __builtin_amdgcn_wmma_f32_16x16x32_bf16(false, ax, false, bR0,
                                                      (short)0, cx0, false, false);
        cm1 = __builtin_amdgcn_wmma_f32_16x16x32_bf16(false, am, false, bL1,
                                                      (short)0, cm1, false, false);
        cx1 = __builtin_amdgcn_wmma_f32_16x16x32_bf16(false, ax, false, bR1,
                                                      (short)0, cx1, false, false);
    }

    __syncthreads();   // all fragment reads done before (possibly in-place) writes

    const float bias0 = bl[n0];
    const float bias1 = bl[n1];
#pragma unroll
    for (int v = 0; v < 8; ++v) {               // D layout: M = v + 8*hi, N = n
        const long row = (long)(m0 + v + hi * 8) * HID;
        out[row + n0] = f2bf(fmaxf(cm0[v] + cx0[v] + bias0, 0.0f));
        out[row + n1] = f2bf(fmaxf(cm1[v] + cx1[v] + bias1, 0.0f));
    }
}

__global__ void pool_scatter(const __bf16* __restrict__ h, const int* __restrict__ batch,
                             float* __restrict__ pooled) {
    long tid = (long)blockIdx.x * blockDim.x + threadIdx.x;
    long node = tid >> 6;
    int  cg = (int)(tid & 63);
    if (node >= N_NODES) return;
    int g = batch[node];
    const v4bf v = *(const v4bf*)(h + node * HID + cg * 4);
    float* p = pooled + (long)g * HID + cg * 4;
    atomicAdd(p + 0, bf2f(v[0]));
    atomicAdd(p + 1, bf2f(v[1]));
    atomicAdd(p + 2, bf2f(v[2]));
    atomicAdd(p + 3, bf2f(v[3]));
}

__global__ void pool_count(const int* __restrict__ batch, float* __restrict__ pcnt) {
    int i = blockIdx.x * blockDim.x + threadIdx.x;
    if (i < N_NODES) atomicAdd(&pcnt[batch[i]], 1.0f);
}

__global__ void head_kernel(const float* __restrict__ pooled, const float* __restrict__ pcnt,
                            const float* __restrict__ Wo, const float* __restrict__ bo,
                            float* __restrict__ out) {
    int g = threadIdx.x;
    if (g >= N_GRAPHS) return;
    float inv = 1.0f / fmaxf(pcnt[g], 1.0f);
    float acc[OUT_CH];
#pragma unroll
    for (int j = 0; j < OUT_CH; ++j) acc[j] = bo[j];
    for (int ch = 0; ch < HID; ++ch) {
        float p = pooled[g * HID + ch] * inv;
#pragma unroll
        for (int j = 0; j < OUT_CH; ++j) acc[j] += p * Wo[ch * OUT_CH + j];
    }
    float mx = acc[0];
#pragma unroll
    for (int j = 1; j < OUT_CH; ++j) mx = fmaxf(mx, acc[j]);
    float s = 0.0f;
#pragma unroll
    for (int j = 0; j < OUT_CH; ++j) s += __expf(acc[j] - mx);
    float ls = __logf(s);
#pragma unroll
    for (int j = 0; j < OUT_CH; ++j) out[g * OUT_CH + j] = acc[j] - mx - ls;
}

extern "C" void kernel_launch(void* const* d_in, const int* in_sizes, int n_in,
                              void* d_out, int out_size, void* d_ws, size_t ws_size,
                              hipStream_t stream) {
    const float* x   = (const float*)d_in[0];
    const int*   ei  = (const int*)d_in[1];
    const int*   bat = (const int*)d_in[2];
    const float* Wl1 = (const float*)d_in[3];
    const float* Wr1 = (const float*)d_in[4];
    const float* bl1 = (const float*)d_in[5];
    const float* Wl  = (const float*)d_in[6];
    const float* Wr  = (const float*)d_in[7];
    const float* bl  = (const float*)d_in[8];
    const float* Wo  = (const float*)d_in[9];
    const float* bo  = (const float*)d_in[10];
    float* outp = (float*)d_out;

    // ---- workspace carve-out (256B aligned bump allocator), ~117 MB ----
    char* ws = (char*)d_ws;
    size_t off = 0;
    auto alloc = [&](size_t bytes) {
        void* p = ws + off;
        off += (bytes + 255) & ~(size_t)255;
        return p;
    };
    float*  aggf   = (float*)alloc((size_t)N_NODES * HID * 4);     // 51.2 MB
    __bf16* hbf    = (__bf16*)alloc((size_t)N_NODES * HID * 2);    // 25.6 MB activations
    __bf16* meanbf = (__bf16*)alloc((size_t)N_NODES * HID * 2);    // 25.6 MB
    __bf16* xbf    = (__bf16*)alloc((size_t)N_NODES * IN_CH * 2);  // 12.8 MB
    __bf16* pWl1   = (__bf16*)alloc((size_t)IN_CH * HID * 2);
    __bf16* pWr1   = (__bf16*)alloc((size_t)IN_CH * HID * 2);
    __bf16* pWls   = (__bf16*)alloc((size_t)4 * HID * HID * 2);
    __bf16* pWrs   = (__bf16*)alloc((size_t)4 * HID * HID * 2);
    float*  rdeg   = (float*)alloc((size_t)N_NODES * 4);
    float*  pooled = (float*)alloc((size_t)N_GRAPHS * HID * 4);
    float*  pcnt   = (float*)alloc((size_t)N_GRAPHS * 4);

    const int MT = N_NODES / 16;               // 3125 exact

    // ---- one-time conversions / weight packing ----
    {
        long nx4 = (long)N_NODES * IN_CH / 4;
        f32_to_bf16_k<<<(int)((nx4 + 255) / 256), 256, 0, stream>>>(x, xbf, nx4);
        pack_w<<<(IN_CH * HID + 255) / 256, 256, 0, stream>>>(Wl1, pWl1, IN_CH);
        pack_w<<<(IN_CH * HID + 255) / 256, 256, 0, stream>>>(Wr1, pWr1, IN_CH);
        for (int l = 0; l < 4; ++l) {
            pack_w<<<(HID * HID + 255) / 256, 256, 0, stream>>>(
                Wl + (long)l * HID * HID, pWls + (long)l * HID * HID, HID);
            pack_w<<<(HID * HID + 255) / 256, 256, 0, stream>>>(
                Wr + (long)l * HID * HID, pWrs + (long)l * HID * HID, HID);
        }
    }

    // ---- reciprocal degrees ----
    zero_f32<<<(N_NODES + 255) / 256, 256, 0, stream>>>(rdeg, N_NODES);
    deg_count<<<(N_EDGES + 255) / 256, 256, 0, stream>>>(ei, rdeg);
    deg_invert<<<(N_NODES + 255) / 256, 256, 0, stream>>>(rdeg);

    // ---- layer 1: 128 -> 256 ----
    {
        long na = (long)N_NODES * IN_CH;
        zero_f32<<<(int)((na + 255) / 256), 256, 0, stream>>>(aggf, na);
        long t = (long)N_EDGES * (IN_CH / 4);
        edge_scatter_f32<<<(int)((t + 255) / 256), 256, 0, stream>>>(ei, x, aggf, rdeg);
        f32_to_bf16_k<<<(int)((na / 4 + 255) / 256), 256, 0, stream>>>(aggf, meanbf, na / 4);
        sage_gemm<IN_CH><<<MT, 256, 0, stream>>>(meanbf, xbf, pWl1, pWr1, bl1, hbf);
    }

    // ---- layers 2..5: 256 -> 256 (GEMM output in place over hbf) ----
    for (int l = 0; l < 4; ++l) {
        long na = (long)N_NODES * HID;
        zero_f32<<<(int)((na + 255) / 256), 256, 0, stream>>>(aggf, na);
        long t = (long)N_EDGES * (HID / 4);
        edge_scatter_bf16<<<(int)((t + 255) / 256), 256, 0, stream>>>(ei, hbf, aggf, rdeg);
        f32_to_bf16_k<<<(int)((na / 4 + 255) / 256), 256, 0, stream>>>(aggf, meanbf, na / 4);
        sage_gemm<HID><<<MT, 256, 0, stream>>>(meanbf, hbf,
                                               pWls + (long)l * HID * HID,
                                               pWrs + (long)l * HID * HID,
                                               bl + l * HID, hbf);
    }

    // ---- global mean pool + head ----
    zero_f32<<<(N_GRAPHS * HID + 255) / 256, 256, 0, stream>>>(pooled, (long)N_GRAPHS * HID);
    zero_f32<<<1, 256, 0, stream>>>(pcnt, N_GRAPHS);
    {
        long t = (long)N_NODES * 64;
        pool_scatter<<<(int)((t + 255) / 256), 256, 0, stream>>>(hbf, bat, pooled);
    }
    pool_count<<<(N_NODES + 255) / 256, 256, 0, stream>>>(bat, pcnt);
    head_kernel<<<1, 128, 0, stream>>>(pooled, pcnt, Wo, bo, outp);
}
